// LSTMModel_54434415509555
// MI455X (gfx1250) — matrix operand order repaired and checked by reading, not compiled, not run on hardware
//
#include <hip/hip_runtime.h>
#include <hip/hip_bf16.h>

// ---------------------------------------------------------------------------
// 2-layer LSTM (B=256,S=512,D=32,H=256) + FC head for MI455X (gfx1250).
// Strategy: bf16 WMMA (v_wmma_f32_16x16x32_bf16) for all matmuls, persistent
// cooperative recurrent kernel (16 WGs, atomic grid barrier, ping-pong h),
// per-WG weight slices cached in LDS, gates i/f/g/o co-located per lane.
// ---------------------------------------------------------------------------

#define B_  256
#define S_  512
#define D_  32
#define H_  256
#define NWG 16

typedef __attribute__((ext_vector_type(16))) __bf16 v16bf;
typedef __attribute__((ext_vector_type(8)))  float  v8f;

// A-matrix (16x32 bf16) per-lane fragment: rows = lane&15; K elements are two
// 8-wide runs at {kbase + half*8, kbase + half*8 + 16}  (half = lane>=16).
__device__ __forceinline__ v16bf ld_a(const __bf16* p) {
  union { uint4 u[2]; v16bf v; } r;
  r.u[0] = *(const uint4*)(p);
  r.u[1] = *(const uint4*)(p + 16);
  return r.v;
}
// B-matrix (32x16 bf16) per-lane fragment: cols = lane&15; K elements are two
// 8-wide runs at {kbase + half*8, kbase + half*8 + 16}  (half = lane>=16).
__device__ __forceinline__ v16bf ld_b(const __bf16* p) {
  union { uint4 u[2]; v16bf v; } r;
  r.u[0] = *(const uint4*)(p);
  r.u[1] = *(const uint4*)(p + 16);
  return r.v;
}

__device__ __forceinline__ float sigf(float x)  { return 1.0f / (1.0f + __expf(-x)); }
__device__ __forceinline__ float tanhf_(float x){ return 2.0f * sigf(2.0f * x) - 1.0f; }

// ---------------------------------------------------------------------------
// Persistent recurrent LSTM layer.
//   xin  : [B, S, Kin] bf16 input sequence (Kin = 32 layer0, 256 layer1)
//   Wih  : [4H, Kin] bf16, Whh : [4H, H] bf16, bias : [4H] f32 (bih+bhh)
//   yout : [B, S, H] bf16 (layer0 only, feeds layer1), may be null
//   hbuf : 2*[B,H] bf16 ping-pong (zeroed), cbuf : [B,H] f32 (zeroed)
//   h_final/c_final : f32 [B,H] slices of d_out (h_n / c_n)
// WG w owns hidden units [16w,16w+16): computes gate columns {q*256+16w+j}.
// Wave v (of 8) owns M-tiles {2v,2v+1}; acc[mt][q] = 16x16 f32 tile.
// ---------------------------------------------------------------------------
__global__ __launch_bounds__(256, 1) void lstm_rec_kernel(
    const __bf16* __restrict__ xin, int Kin,
    const __bf16* __restrict__ Wih, const __bf16* __restrict__ Whh,
    const float*  __restrict__ bias,
    __bf16* __restrict__ yout,
    __bf16* __restrict__ hbuf,
    float*  __restrict__ cbuf,
    float*  __restrict__ h_final, float* __restrict__ c_final,
    unsigned* __restrict__ bar_cnt, unsigned* __restrict__ bar_gen)
{
  extern __shared__ __bf16 smem[];
  __bf16* sWhh = smem;              // 64 rows x 256 (this WG's Whh slice)
  __bf16* sWih = smem + 64 * 256;   // 64 rows x Kin

  const int wg   = blockIdx.x;      // 0..15
  const int tid  = threadIdx.x;
  const int lane = tid & 31;
  const int wv   = tid >> 5;        // 0..7
  const int col  = lane & 15;
  const int half = lane >> 4;       // 0/1

  // Stage this WG's weight rows into LDS (gate row = q*256 + wg*16 + j).
  for (int i = tid; i < 64 * 256; i += 256) {
    int rl = i >> 8, k = i & 255;
    int grow = ((rl >> 4) << 8) + wg * 16 + (rl & 15);
    sWhh[i] = Whh[(size_t)grow * 256 + k];
  }
  for (int i = tid; i < 64 * Kin; i += 256) {
    int rl = i / Kin, k = i - rl * Kin;
    int grow = ((rl >> 4) << 8) + wg * 16 + (rl & 15);
    sWih[i] = Wih[(size_t)grow * Kin + k];
  }
  __syncthreads();

  float bval[4];
  #pragma unroll
  for (int q = 0; q < 4; ++q) bval[q] = bias[q * 256 + wg * 16 + col];

  const int kstep_in = Kin >> 5;
  const int ka = half * 8;    // A fragment K base offset
  const int kb = half * 8;    // B fragment K base offset
  const int u  = wg * 16 + col;

  for (int t = 0; t < S_; ++t) {
    const __bf16* hr = hbuf + (size_t)(t & 1) * (B_ * H_);
    __bf16*       hw = hbuf + (size_t)((t + 1) & 1) * (B_ * H_);

    v8f acc[2][4];
    #pragma unroll
    for (int mt = 0; mt < 2; ++mt)
      #pragma unroll
      for (int q = 0; q < 4; ++q) {
        v8f z;
        #pragma unroll
        for (int r = 0; r < 8; ++r) z[r] = bval[q];
        acc[mt][q] = z;
      }

    // Input projection: gates += x_t @ Wih^T   (K = Kin)
    for (int kt = 0; kt < kstep_in; ++kt) {
      v16bf a[2];
      #pragma unroll
      for (int mt = 0; mt < 2; ++mt) {
        int m = (wv * 2 + mt) * 16 + col;
        a[mt] = ld_a(xin + ((size_t)m * S_ + t) * Kin + kt * 32 + ka);
      }
      #pragma unroll
      for (int q = 0; q < 4; ++q) {
        v16bf b = ld_b(sWih + (size_t)(q * 16 + col) * Kin + kt * 32 + kb);
        #pragma unroll
        for (int mt = 0; mt < 2; ++mt)
          acc[mt][q] = __builtin_amdgcn_wmma_f32_16x16x32_bf16(
              false, a[mt], false, b, (short)0, acc[mt][q], false, false);
      }
    }

    // Recurrent: gates += h @ Whh^T   (K = 256)
    #pragma unroll
    for (int kt = 0; kt < 8; ++kt) {
      v16bf a[2];
      #pragma unroll
      for (int mt = 0; mt < 2; ++mt) {
        int m = (wv * 2 + mt) * 16 + col;
        a[mt] = ld_a(hr + (size_t)m * H_ + kt * 32 + ka);
      }
      #pragma unroll
      for (int q = 0; q < 4; ++q) {
        v16bf b = ld_b(sWhh + (size_t)(q * 16 + col) * 256 + kt * 32 + kb);
        #pragma unroll
        for (int mt = 0; mt < 2; ++mt)
          acc[mt][q] = __builtin_amdgcn_wmma_f32_16x16x32_bf16(
              false, a[mt], false, b, (short)0, acc[mt][q], false, false);
      }
    }

    // Cell update: lane holds i,f,g,o for the same (row m, unit u).
    #pragma unroll
    for (int mt = 0; mt < 2; ++mt) {
      #pragma unroll
      for (int r = 0; r < 8; ++r) {
        int m = (wv * 2 + mt) * 16 + half * 8 + r;   // C/D row mapping
        size_t idx = (size_t)m * H_ + u;
        float i_ = sigf(acc[mt][0][r]);
        float f_ = sigf(acc[mt][1][r]);
        float g_ = tanhf_(acc[mt][2][r]);
        float o_ = sigf(acc[mt][3][r]);
        float cn = f_ * cbuf[idx] + i_ * g_;
        float hn = o_ * tanhf_(cn);
        cbuf[idx] = cn;
        hw[idx]   = (__bf16)hn;
        if (yout) yout[((size_t)m * S_ + t) * H_ + u] = (__bf16)hn;
        if (t == S_ - 1) { h_final[idx] = hn; c_final[idx] = cn; }
      }
    }

    // Grid-wide barrier (release new h to all 16 WGs).
    __syncthreads();
    if (tid == 0) {
      __threadfence();
      if (atomicAdd(bar_cnt, 1u) == NWG - 1u) {
        *bar_cnt = 0u;
        __threadfence();
        atomicAdd(bar_gen, 1u);
      } else {
        volatile unsigned* vg = bar_gen;
        while (*vg <= (unsigned)t) __builtin_amdgcn_s_sleep(1);
      }
      __threadfence();
    }
    __syncthreads();
  }
}

// hid = relu(last @ W1^T + b1); out = hid @ W2^T + b2. One block per batch row.
__global__ __launch_bounds__(256) void fc_kernel(
    const float* __restrict__ last, const float* __restrict__ W1,
    const float* __restrict__ b1,   const float* __restrict__ W2,
    const float* __restrict__ b2,   float* __restrict__ out)
{
  __shared__ float sh[256];
  int b = blockIdx.x, j = threadIdx.x;
  const float* row = last + (size_t)b * H_;
  float s = b1[j];
  for (int k = 0; k < H_; ++k) s = fmaf(row[k], W1[(size_t)j * H_ + k], s);
  sh[j] = fmaxf(s, 0.0f) * W2[j];
  __syncthreads();
  for (int o = 128; o > 0; o >>= 1) {
    if (j < o) sh[j] += sh[j + o];
    __syncthreads();
  }
  if (j == 0) out[b] = sh[0] + b2[0];
}

__global__ void cast_bf16_kernel(const float* __restrict__ s,
                                 __bf16* __restrict__ d, size_t n) {
  size_t i = (size_t)blockIdx.x * blockDim.x + threadIdx.x;
  size_t st = (size_t)gridDim.x * blockDim.x;
  for (; i < n; i += st) d[i] = (__bf16)s[i];
}

__global__ void bias_sum_kernel(const float* __restrict__ a,
                                const float* __restrict__ b,
                                float* __restrict__ d, int n) {
  int i = blockIdx.x * blockDim.x + threadIdx.x;
  if (i < n) d[i] = a[i] + b[i];
}

__global__ void zero_u32_kernel(unsigned* __restrict__ p, size_t nwords) {
  size_t i = (size_t)blockIdx.x * blockDim.x + threadIdx.x;
  size_t st = (size_t)gridDim.x * blockDim.x;
  for (; i < nwords; i += st) p[i] = 0u;
}

extern "C" void kernel_launch(void* const* d_in, const int* in_sizes, int n_in,
                              void* d_out, int out_size, void* d_ws, size_t ws_size,
                              hipStream_t stream) {
  (void)in_sizes; (void)n_in; (void)out_size; (void)ws_size;
  const float* x    = (const float*)d_in[0];
  const float* Wih0 = (const float*)d_in[1];
  const float* Whh0 = (const float*)d_in[2];
  const float* bih0 = (const float*)d_in[3];
  const float* bhh0 = (const float*)d_in[4];
  const float* Wih1 = (const float*)d_in[5];
  const float* Whh1 = (const float*)d_in[6];
  const float* bih1 = (const float*)d_in[7];
  const float* bhh1 = (const float*)d_in[8];
  const float* fcW1 = (const float*)d_in[9];
  const float* fcb1 = (const float*)d_in[10];
  const float* fcW2 = (const float*)d_in[11];
  const float* fcb2 = (const float*)d_in[12];

  char* ws = (char*)d_ws;
  size_t off = 0;
  auto alloc = [&](size_t bytes) -> void* {
    void* p = ws + off;
    off = (off + bytes + 255) & ~(size_t)255;
    return p;
  };

  __bf16* x_bf  = (__bf16*)alloc((size_t)B_ * S_ * D_ * 2);   // 8 MB
  __bf16* y0_bf = (__bf16*)alloc((size_t)B_ * S_ * H_ * 2);   // 64 MB
  __bf16* wih0b = (__bf16*)alloc((size_t)4 * H_ * D_ * 2);
  __bf16* whh0b = (__bf16*)alloc((size_t)4 * H_ * H_ * 2);
  __bf16* wih1b = (__bf16*)alloc((size_t)4 * H_ * H_ * 2);
  __bf16* whh1b = (__bf16*)alloc((size_t)4 * H_ * H_ * 2);
  float*  b0    = (float*)alloc((size_t)4 * H_ * 4);
  float*  b1    = (float*)alloc((size_t)4 * H_ * 4);
  size_t state_beg = off;
  __bf16* h0buf = (__bf16*)alloc((size_t)2 * B_ * H_ * 2);    // ping-pong
  __bf16* h1buf = (__bf16*)alloc((size_t)2 * B_ * H_ * 2);
  float*  c0buf = (float*)alloc((size_t)B_ * H_ * 4);
  float*  c1buf = (float*)alloc((size_t)B_ * H_ * 4);
  unsigned* bar = (unsigned*)alloc(8 * sizeof(unsigned));
  size_t state_end = off;

  float* out   = (float*)d_out;                 // [B,1]
  float* h_n0  = out + B_;                      // h_n[0] : [B,H]
  float* h_n1  = h_n0 + (size_t)B_ * H_;        // h_n[1]
  float* c_n0  = h_n1 + (size_t)B_ * H_;        // c_n[0]
  float* c_n1  = c_n0 + (size_t)B_ * H_;        // c_n[1]

  // ---- prep: bf16 casts, combined biases, zero state + barriers ----
  {
    size_t n = (size_t)B_ * S_ * D_;
    cast_bf16_kernel<<<(unsigned)((n + 255) / 256), 256, 0, stream>>>(x, x_bf, n);
  }
  cast_bf16_kernel<<<(4 * H_ * D_) / 256, 256, 0, stream>>>(Wih0, wih0b, (size_t)4 * H_ * D_);
  cast_bf16_kernel<<<(4 * H_ * H_) / 256, 256, 0, stream>>>(Whh0, whh0b, (size_t)4 * H_ * H_);
  cast_bf16_kernel<<<(4 * H_ * H_) / 256, 256, 0, stream>>>(Wih1, wih1b, (size_t)4 * H_ * H_);
  cast_bf16_kernel<<<(4 * H_ * H_) / 256, 256, 0, stream>>>(Whh1, whh1b, (size_t)4 * H_ * H_);
  bias_sum_kernel<<<(4 * H_) / 256, 256, 0, stream>>>(bih0, bhh0, b0, 4 * H_);
  bias_sum_kernel<<<(4 * H_) / 256, 256, 0, stream>>>(bih1, bhh1, b1, 4 * H_);
  {
    size_t nwords = (state_end - state_beg) / 4;
    zero_u32_kernel<<<1024, 256, 0, stream>>>((unsigned*)(ws + state_beg), nwords);
  }

  // ---- layer 0: persistent recurrent (Kin = 32) ----
  size_t shmem0 = (size_t)(64 * 256 + 64 * D_) * 2;
  lstm_rec_kernel<<<NWG, 256, shmem0, stream>>>(
      x_bf, D_, wih0b, whh0b, b0, y0_bf, h0buf, c0buf, h_n0, c_n0,
      bar + 0, bar + 1);

  // ---- layer 1: persistent recurrent, input proj fused (Kin = 256) ----
  size_t shmem1 = (size_t)(64 * 256 + 64 * H_) * 2;
  lstm_rec_kernel<<<NWG, 256, shmem1, stream>>>(
      y0_bf, H_, wih1b, whh1b, b1, nullptr, h1buf, c1buf, h_n1, c_n1,
      bar + 2, bar + 3);

  // ---- FC head: last = y1[:, -1, :] == h_n[1] ----
  fc_kernel<<<B_, 256, 0, stream>>>(h_n1, fcW1, fcb1, fcW2, fcb2, out);
}
